// DualRGATLayer_2224793059900
// MI455X (gfx1250) — compile-verified, matrix operand order checked by reading, not run on hardware
//
#include <hip/hip_runtime.h>

// ---------------------------------------------------------------------------
// DualRGATLayer for MI455X (gfx1250): bf16 WMMA GEMMs + atomic segment attention
// N=10000 nodes, E=160000 edges, ELG=320000 line-graph edges, D=256, H=8, DK=32
//
// GEMMs use pre-transposed bf16 weights (Bt[n][k]); A/B tiles are staged into
// double-buffered LDS via GLOBAL_LOAD_ASYNC_TO_LDS_B128 (ASYNCcnt) when the
// toolchain exposes the builtin, else via b128 VGPR staging. 4 WMMAs per wave
// per K-step. All launches on `stream`; workspace bump-allocated from d_ws.
// ---------------------------------------------------------------------------

typedef unsigned short u16;
typedef __bf16 bf16_t;
typedef bf16_t bf16x16 __attribute__((ext_vector_type(16)));
typedef float  floatx8 __attribute__((ext_vector_type(8)));

#define BM 64
#define BN 128
#define BK 32
#define LDST 40   // LDS row stride in halves (32 data + 8 pad -> 80B rows, 16B aligned)

#if defined(__has_builtin)
#if __has_builtin(__builtin_amdgcn_global_load_async_to_lds_b128)
#define USE_ASYNC_LDS 1
#endif
#endif
#ifndef USE_ASYNC_LDS
#define USE_ASYNC_LDS 0
#endif

#if USE_ASYNC_LDS
// builtin expects pointers to 16-byte int vectors (per hipcc diagnostic)
typedef int i32x4v __attribute__((vector_size(4 * sizeof(int))));
typedef __attribute__((address_space(1))) i32x4v g_i32x4;
typedef __attribute__((address_space(3))) i32x4v l_i32x4;
#endif

__device__ __forceinline__ u16 f2bf(float f) {
    unsigned int u = __float_as_uint(f);
    u += 0x7FFFu + ((u >> 16) & 1u);   // round to nearest even
    return (u16)(u >> 16);
}

// 16-byte global -> LDS copy: async DMA path (ASYNCcnt) or VGPR fallback
__device__ __forceinline__ void copy16_g2l(const u16* __restrict__ g, u16* l) {
#if USE_ASYNC_LDS
    __builtin_amdgcn_global_load_async_to_lds_b128((g_i32x4*)(const void*)g,
                                                   (l_i32x4*)(void*)l, 0, 0);
#else
    *(uint4*)l = *(const uint4*)g;
#endif
}

__device__ __forceinline__ void async_wait_all() {
#if USE_ASYNC_LDS
    asm volatile("s_wait_asynccnt 0x0" ::: "memory");
#endif
}

union FragU { uint4 u[2]; bf16x16 v; };

// Build a 16x32 bf16 fragment. Lanes 0-15: row=lane, K halves {0..7,16..23};
// lanes 16-31: row=lane-16, K halves {8..15,24..31}. Same layout for A rows
// and (pre-transposed) B columns.
__device__ __forceinline__ bf16x16 frag_from_lds(const u16* lds, int rowbase, int lane) {
    int r = rowbase + (lane & 15);
    int ho = (lane & 16) ? 8 : 0;
    FragU f;
    f.u[0] = *(const uint4*)(lds + r * LDST + ho);
    f.u[1] = *(const uint4*)(lds + r * LDST + 16 + ho);
    return f.v;
}

// C[M,Ncol] = A[M,K](bf16) @ Bt[Ncol,K](bf16, pre-transposed)
//             (+bias) (+res) (+gath[gidx[row]]) (relu) -> f32 or bf16
template<bool RELU, bool OUT_BF16>
__global__ __launch_bounds__(256, 2)
void gemm_bf16_wmma(const u16* __restrict__ A, const u16* __restrict__ Bt,
                    float* __restrict__ Cf, u16* __restrict__ Ch,
                    const float* __restrict__ bias, const float* __restrict__ res,
                    const float* __restrict__ gath, const int* __restrict__ gidx,
                    int M, int K, int Ncol) {
    __shared__ __attribute__((aligned(16))) u16 lA[2][BM * LDST];
    __shared__ __attribute__((aligned(16))) u16 lB[2][BN * LDST];

    const int tid  = threadIdx.x;
    const int lane = tid & 31;
    const int wave = tid >> 5;
    const int wm   = (wave & 1) * 32;   // 2 waves along M
    const int wn   = (wave >> 1) * 32;  // 4 waves along N
    const int m0   = blockIdx.x * BM;
    const int n0   = blockIdx.y * BN;

    // per-thread staging coordinates: 16B chunks of 8 halves
    const int ra = tid >> 2, ca = tid & 3;   // A: 64 rows x 4 chunks
    const int rb = tid >> 2, cb = tid & 3;   // B: rows rb and rb+64, 4 chunks

    const floatx8 zv = {0.f,0.f,0.f,0.f,0.f,0.f,0.f,0.f};
    floatx8 acc[2][2];
    acc[0][0] = zv; acc[0][1] = zv; acc[1][0] = zv; acc[1][1] = zv;

    auto stage = [&](int buf, int k0) {
        int gr = m0 + ra;
        u16* la = &lA[buf][ra * LDST + ca * 8];
        if (gr < M) {
            copy16_g2l(A + (size_t)gr * K + k0 + ca * 8, la);
        } else {
            *(uint4*)la = make_uint4(0u, 0u, 0u, 0u);   // M-tail zero fill
        }
        copy16_g2l(Bt + (size_t)(n0 + rb) * K + k0 + cb * 8,
                   &lB[buf][rb * LDST + cb * 8]);
        copy16_g2l(Bt + (size_t)(n0 + rb + 64) * K + k0 + cb * 8,
                   &lB[buf][(rb + 64) * LDST + cb * 8]);
    };

    const int nk = K / BK;
    stage(0, 0);
    async_wait_all();
    __syncthreads();

    for (int it = 0; it < nk; ++it) {
        const int cur = it & 1;
        if (it + 1 < nk) {
            if (it + 2 < nk && m0 + ra < M) {
                // pull the K+2 activation tile toward the caches (global_prefetch_b8)
                __builtin_prefetch(A + (size_t)(m0 + ra) * K + (it + 2) * BK + ca * 8, 0, 1);
            }
            stage(cur ^ 1, (it + 1) * BK);   // overlaps with WMMAs below
        }
        bf16x16 a0 = frag_from_lds(lA[cur], wm,      lane);
        bf16x16 a1 = frag_from_lds(lA[cur], wm + 16, lane);
        bf16x16 b0 = frag_from_lds(lB[cur], wn,      lane);
        bf16x16 b1 = frag_from_lds(lB[cur], wn + 16, lane);

        acc[0][0] = __builtin_amdgcn_wmma_f32_16x16x32_bf16(false, a0, false, b0, (short)0, acc[0][0], false, false);
        acc[0][1] = __builtin_amdgcn_wmma_f32_16x16x32_bf16(false, a0, false, b1, (short)0, acc[0][1], false, false);
        acc[1][0] = __builtin_amdgcn_wmma_f32_16x16x32_bf16(false, a1, false, b0, (short)0, acc[1][0], false, false);
        acc[1][1] = __builtin_amdgcn_wmma_f32_16x16x32_bf16(false, a1, false, b1, (short)0, acc[1][1], false, false);

        async_wait_all();    // own async copies for the next tile have landed
        __syncthreads();     // everyone's copies landed / reads of cur done
    }

    // ---- epilogue + store: VGPR v holds M=v (lanes 0-15) / M=8+v (lanes 16-31)
#pragma unroll
    for (int i = 0; i < 2; ++i) {
        int rowb = m0 + wm + i * 16 + ((lane & 16) ? 8 : 0);
#pragma unroll
        for (int j = 0; j < 2; ++j) {
            int col = n0 + wn + j * 16 + (lane & 15);
            float bval = bias ? bias[col] : 0.0f;
#pragma unroll
            for (int vv = 0; vv < 8; ++vv) {
                int row = rowb + vv;
                if (row < M) {
                    float val = acc[i][j][vv] + bval;
                    if (res)  val += res[(size_t)row * Ncol + col];
                    if (gath) val += gath[(size_t)gidx[row] * Ncol + col];
                    if (RELU) val = fmaxf(val, 0.0f);
                    if (OUT_BF16) Ch[(size_t)row * Ncol + col] = f2bf(val);
                    else          Cf[(size_t)row * Ncol + col] = val;
                }
            }
        }
    }
}

// f32 -> bf16 conversion (activations)
__global__ void cvt_bf16_kernel(const float* __restrict__ in, u16* __restrict__ out, long n) {
    long t = (long)blockIdx.x * blockDim.x + threadIdx.x;
    if (t < n) out[t] = f2bf(in[t]);
}

// f32 W[K][Ncol] -> bf16 Bt[Ncol][K] (one-time weight transpose)
__global__ void cvtT_bf16_kernel(const float* __restrict__ in, u16* __restrict__ out,
                                 int K, int Ncol) {
    long t = (long)blockIdx.x * blockDim.x + threadIdx.x;
    if (t < (long)K * Ncol) {
        int k = (int)(t / Ncol), n = (int)(t % Ncol);
        out[(size_t)n * K + k] = f2bf(in[t]);
    }
}

__global__ void zero_kernel(float* __restrict__ p, long n) {
    long t = (long)blockIdx.x * blockDim.x + threadIdx.x;
    if (t < n) p[t] = 0.0f;
}

// Segment attention: wave per edge, lane per DK dim, loop over 8 heads.
// score = exp(clamp(dot(k[src]+e, q[dst]) / sqrt(32), -10, 10))
// wv[dst] += (v[src]+e) * score ; z[dst] += score   (atomics)
__global__ __launch_bounds__(256)
void attn_kernel(const float* __restrict__ q, const float* __restrict__ k,
                 const float* __restrict__ v, const float* __restrict__ e,
                 const int* __restrict__ src, const int* __restrict__ dst,
                 float* __restrict__ wv, float* __restrict__ z, int Eg) {
    const int wave = threadIdx.x >> 5;
    const int lane = threadIdx.x & 31;
    const int edge = blockIdx.x * 8 + wave;
    if (edge >= Eg) return;
    const int s = src[edge], d = dst[edge];
    const float* kr = k + (size_t)s * 256;
    const float* qr = q + (size_t)d * 256;
    const float* vr = v + (size_t)s * 256;
    const float* er = e ? (e + (size_t)edge * 256) : nullptr;
#pragma unroll
    for (int h = 0; h < 8; ++h) {
        int c = h * 32 + lane;
        float ev = er ? er[c] : 0.0f;
        float p = (kr[c] + ev) * qr[c];
#pragma unroll
        for (int m = 16; m; m >>= 1) p += __shfl_xor(p, m);   // wave32 reduce
        float sc = p * 0.17677669529663687f;                   // 1/sqrt(32)
        sc = __expf(fminf(10.0f, fmaxf(-10.0f, sc)));
        atomicAdd(&wv[(size_t)d * 256 + c], (vr[c] + ev) * sc);
        if (lane == 0) atomicAdd(&z[(size_t)d * 8 + h], sc);
    }
}

// o = wv / z (per head), emitted directly as bf16 for the Wo GEMM
__global__ void attn_norm_kernel(const float* __restrict__ wv, const float* __restrict__ z,
                                 u16* __restrict__ out, long M) {
    long t = (long)blockIdx.x * blockDim.x + threadIdx.x;
    if (t < M * 256) {
        long row = t >> 8;
        int  h   = (int)((t >> 5) & 7);
        out[t] = f2bf(wv[t] / z[row * 8 + h]);
    }
}

// LayerNorm over D=256: wave per row, 8 cols per lane; dual f32 + bf16 output
__global__ __launch_bounds__(256)
void ln_kernel(const float* __restrict__ in, const float* __restrict__ g,
               const float* __restrict__ b, float* __restrict__ outf,
               u16* __restrict__ outh, int M) {
    const int wave = threadIdx.x >> 5;
    const int lane = threadIdx.x & 31;
    const int row  = blockIdx.x * 8 + wave;
    if (row >= M) return;
    const float* r = in + (size_t)row * 256;
    float vals[8];
    float s = 0.0f;
#pragma unroll
    for (int i = 0; i < 8; ++i) { vals[i] = r[lane + 32 * i]; s += vals[i]; }
#pragma unroll
    for (int m = 16; m; m >>= 1) s += __shfl_xor(s, m);
    float mean = s * (1.0f / 256.0f);
    float sq = 0.0f;
#pragma unroll
    for (int i = 0; i < 8; ++i) { float d = vals[i] - mean; sq += d * d; }
#pragma unroll
    for (int m = 16; m; m >>= 1) sq += __shfl_xor(sq, m);
    float rstd = rsqrtf(sq * (1.0f / 256.0f) + 1e-5f);
#pragma unroll
    for (int i = 0; i < 8; ++i) {
        int c = lane + 32 * i;
        float o = (vals[i] - mean) * rstd * g[c] + b[c];
        if (outf) outf[(size_t)row * 256 + c] = o;
        if (outh) outh[(size_t)row * 256 + c] = f2bf(o);
    }
}

// ---------------------------------------------------------------------------

extern "C" void kernel_launch(void* const* d_in, const int* in_sizes, int n_in,
                              void* d_out, int out_size, void* d_ws, size_t ws_size,
                              hipStream_t stream) {
    (void)in_sizes; (void)n_in; (void)out_size; (void)ws_size;
    const int N = 10000, E = 160000, ELG = 320000, D = 256, H = 8;
    const int D4 = 4 * D;

    auto F = [&](int i) { return (const float*)d_in[i]; };
    auto I = [&](int i) { return (const int*)d_in[i]; };

    const float* x         = F(0);
    const float* lgx       = F(1);   // local_lgx (global_lgx d_in[2] unused, per reference)
    const int*   local_src = I(3);
    const int*   local_dst = I(4);
    const int*   lg_src    = I(5);
    const int*   lg_dst    = I(6);
    const int*   src_ids   = I(7);
    const int*   dst_ids   = I(8);
    // node params 9..22, edge params 23..36
    const float *nWq=F(9),  *nbq=F(10), *nWk=F(11), *nWv=F(12), *nWo=F(13), *nbo=F(14);
    const float *nl1g=F(15),*nl1b=F(16),*nW1=F(17), *nb1=F(18), *nW2=F(19), *nb2=F(20);
    const float *nl2g=F(21),*nl2b=F(22);
    const float *eWq=F(23), *ebq=F(24), *eWk=F(25), *eWv=F(26), *eWo=F(27), *ebo=F(28);
    const float *el1g=F(29),*el1b=F(30),*eW1=F(31), *eb1=F(32), *eW2=F(33), *eb2=F(34);
    const float *el2g=F(35),*el2b=F(36);

    // ---- workspace bump allocator ----
    char* wsp = (char*)d_ws;
    size_t off = 0;
    auto alloc = [&](size_t bytes) -> void* {
        void* p = wsp + off;
        off += (bytes + 255) & ~(size_t)255;
        return p;
    };
    // bf16 weights, pre-transposed to [Ncol][K]
    u16 *nWqb=(u16*)alloc((size_t)D*D*2),  *nWkb=(u16*)alloc((size_t)D*D*2);
    u16 *nWvb=(u16*)alloc((size_t)D*D*2),  *nWob=(u16*)alloc((size_t)D*D*2);
    u16 *nW1b=(u16*)alloc((size_t)D*D4*2), *nW2b=(u16*)alloc((size_t)D4*D*2);
    u16 *eWqb=(u16*)alloc((size_t)D*D*2),  *eWkb=(u16*)alloc((size_t)D*D*2);
    u16 *eWvb=(u16*)alloc((size_t)D*D*2),  *eWob=(u16*)alloc((size_t)D*D*2);
    u16 *eW1b=(u16*)alloc((size_t)D*D4*2), *eW2b=(u16*)alloc((size_t)D4*D*2);
    // bf16 activations
    u16 *xb   = (u16*)alloc((size_t)N*D*2);
    u16 *lgxb = (u16*)alloc((size_t)E*D*2);
    // node-side intermediates (small)
    float *q    = (float*)alloc((size_t)N*D*4);
    float *kk   = (float*)alloc((size_t)N*D*4);
    float *v    = (float*)alloc((size_t)N*D*4);
    float *wv   = (float*)alloc((size_t)N*D*4);
    float *z    = (float*)alloc((size_t)N*H*4);
    u16   *ob   = (u16*)  alloc((size_t)N*D*2);
    float *pre1 = (float*)alloc((size_t)N*D*4);
    float *hbuf = (float*)alloc((size_t)N*D*4);
    u16   *hb   = (u16*)  alloc((size_t)N*D*2);
    u16   *f1b  = (u16*)  alloc((size_t)N*D4*2);
    float *pre2 = (float*)alloc((size_t)N*D*4);
    // edge-side (large; aliased where lifetimes allow)
    float *qe   = (float*)alloc((size_t)E*D*4);
    float *ke   = (float*)alloc((size_t)E*D*4);
    float *ve   = (float*)alloc((size_t)E*D*4);
    float *wve  = (float*)alloc((size_t)E*D*4);
    float *ze   = (float*)alloc((size_t)E*H*4);
    u16   *oeb  = (u16*)  alloc((size_t)E*D*2);
    u16   *heb  = (u16*)  alloc((size_t)E*D*2);
    u16   *f1eb = (u16*)  alloc((size_t)E*D4*2);
    float *pre1e = ke;   // ke dead after edge attention
    float *hebuf = ve;   // ve dead after edge attention
    float *pre2e = qe;   // qe dead after edge attention

    float* out_x   = (float*)d_out;
    float* out_lgx = (float*)d_out + (size_t)N * D;

    auto cdiv = [](long a, long b) { return (int)((a + b - 1) / b); };
    auto cvt = [&](const float* s, u16* d, long n) {
        cvt_bf16_kernel<<<cdiv(n, 256), 256, 0, stream>>>(s, d, n);
    };
    auto cvtT = [&](const float* s, u16* d, int K, int Ncol) {
        cvtT_bf16_kernel<<<cdiv((long)K * Ncol, 256), 256, 0, stream>>>(s, d, K, Ncol);
    };

    // ---- stage 0: bf16 conversions (weights transposed once) ----
    cvtT(nWq, nWqb, D, D);   cvtT(nWk, nWkb, D, D);
    cvtT(nWv, nWvb, D, D);   cvtT(nWo, nWob, D, D);
    cvtT(nW1, nW1b, D, D4);  cvtT(nW2, nW2b, D4, D);
    cvtT(eWq, eWqb, D, D);   cvtT(eWk, eWkb, D, D);
    cvtT(eWv, eWvb, D, D);   cvtT(eWo, eWob, D, D);
    cvtT(eW1, eW1b, D, D4);  cvtT(eW2, eW2b, D4, D);
    cvt(x, xb, (long)N*D);
    cvt(lgx, lgxb, (long)E*D);

    dim3 blk(256);
    // ================= node update =================
    {
        dim3 g(cdiv(N, BM), D / BN);
        gemm_bf16_wmma<false,false><<<g, blk, 0, stream>>>(xb, nWqb, q,  nullptr, nbq,    nullptr, nullptr, nullptr, N, D, D);
        gemm_bf16_wmma<false,false><<<g, blk, 0, stream>>>(xb, nWkb, kk, nullptr, nullptr,nullptr, nullptr, nullptr, N, D, D);
        gemm_bf16_wmma<false,false><<<g, blk, 0, stream>>>(xb, nWvb, v,  nullptr, nullptr,nullptr, nullptr, nullptr, N, D, D);
    }
    zero_kernel<<<cdiv((long)N*D, 256), 256, 0, stream>>>(wv, (long)N*D);
    zero_kernel<<<cdiv((long)N*H, 256), 256, 0, stream>>>(z,  (long)N*H);
    attn_kernel<<<cdiv(E, 8), 256, 0, stream>>>(q, kk, v, lgx, local_src, local_dst, wv, z, E);
    attn_norm_kernel<<<cdiv((long)N*D, 256), 256, 0, stream>>>(wv, z, ob, N);
    {
        dim3 g(cdiv(N, BM), D / BN);
        gemm_bf16_wmma<false,false><<<g, blk, 0, stream>>>(ob, nWob, pre1, nullptr, nbo, x, nullptr, nullptr, N, D, D);
    }
    ln_kernel<<<cdiv(N, 8), 256, 0, stream>>>(pre1, nl1g, nl1b, hbuf, hb, N);
    {
        dim3 g1(cdiv(N, BM), D4 / BN);
        gemm_bf16_wmma<true,true><<<g1, blk, 0, stream>>>(hb, nW1b, nullptr, f1b, nb1, nullptr, nullptr, nullptr, N, D, D4);
        dim3 g2(cdiv(N, BM), D / BN);
        gemm_bf16_wmma<false,false><<<g2, blk, 0, stream>>>(f1b, nW2b, pre2, nullptr, nb2, hbuf, nullptr, nullptr, N, D4, D);
    }
    ln_kernel<<<cdiv(N, 8), 256, 0, stream>>>(pre2, nl2g, nl2b, out_x, nullptr, N);

    // ================= edge update (line graph) =================
    {
        dim3 g(cdiv(E, BM), D / BN);
        gemm_bf16_wmma<false,false><<<g, blk, 0, stream>>>(lgxb, eWqb, qe, nullptr, ebq,     nullptr, x, src_ids, E, D, D);
        gemm_bf16_wmma<false,false><<<g, blk, 0, stream>>>(lgxb, eWkb, ke, nullptr, nullptr, nullptr, nullptr, nullptr, E, D, D);
        gemm_bf16_wmma<false,false><<<g, blk, 0, stream>>>(lgxb, eWvb, ve, nullptr, nullptr, nullptr, x, dst_ids, E, D, D);
    }
    zero_kernel<<<cdiv((long)E*D, 256), 256, 0, stream>>>(wve, (long)E*D);
    zero_kernel<<<cdiv((long)E*H, 256), 256, 0, stream>>>(ze,  (long)E*H);
    attn_kernel<<<cdiv(ELG, 8), 256, 0, stream>>>(qe, ke, ve, nullptr, lg_src, lg_dst, wve, ze, ELG);
    attn_norm_kernel<<<cdiv((long)E*D, 256), 256, 0, stream>>>(wve, ze, oeb, E);
    {
        dim3 g(cdiv(E, BM), D / BN);
        gemm_bf16_wmma<false,false><<<g, blk, 0, stream>>>(oeb, eWob, pre1e, nullptr, ebo, lgx, nullptr, nullptr, E, D, D);
    }
    ln_kernel<<<cdiv(E, 8), 256, 0, stream>>>(pre1e, el1g, el1b, hebuf, heb, E);
    {
        dim3 g1(cdiv(E, BM), D4 / BN);
        gemm_bf16_wmma<true,true><<<g1, blk, 0, stream>>>(heb, eW1b, nullptr, f1eb, eb1, nullptr, nullptr, nullptr, E, D, D4);
        dim3 g2(cdiv(E, BM), D / BN);
        gemm_bf16_wmma<false,false><<<g2, blk, 0, stream>>>(f1eb, eW2b, pre2e, nullptr, eb2, hebuf, nullptr, nullptr, E, D4, D);
    }
    ln_kernel<<<cdiv(E, 8), 256, 0, stream>>>(pre2e, el2g, el2b, out_lgx, nullptr, E);
}
